// PointNet2_47090021433719
// MI455X (gfx1250) — compile-verified, matrix operand order checked by reading, not run on hardware
//
#include <hip/hip_runtime.h>

// ---------------------------------------------------------------------------
// CDNA5 (gfx1250) PointNet++ pipeline.  wave32, WMMA f32<-f16 16x16x32.
// All GEMM operands pre-packed to f16, K-contiguous, so every fragment load
// is two 16-byte vector loads (global_load_b128 / ds_load_b128).
// ---------------------------------------------------------------------------

typedef __attribute__((ext_vector_type(16))) _Float16 v16h;
typedef __attribute__((ext_vector_type(8)))  float    v8f;

#define BIGF 1.0e9f

// K index inside a 32-wide K block for the 16-bit A/B WMMA fragment layout:
// VGPR v (element pair e>>1): K base = (v<4 ? 2v : 2v+8), +8 for lane-half 1.
// Consequence: elements e=0..7 cover K = kbase+8*half .. +7 (contiguous),
//              elements e=8..15 cover K = kbase+8*half+16 .. +7 (contiguous).
__device__ __forceinline__ int kmap(int e, int half) {
    int v = e >> 1;
    return ((v < 4) ? 2 * v : 2 * v + 8) + 8 * half + (e & 1);
}

__device__ __forceinline__ v8f wmma_f16(v16h a, v16h b, v8f c) {
    return __builtin_amdgcn_wmma_f32_16x16x32_f16(
        /*neg_a=*/false, a, /*neg_b=*/false, b,
        /*c_mod=*/(short)0, c, /*reuse_a=*/false, /*reuse_b=*/false);
}

// Fragment load from an f16 K-contiguous row (global or LDS), 16B aligned.
// Two uint4 loads at +0 and +32 bytes give the exact v16h element order.
__device__ __forceinline__ v16h load_frag16(const _Float16* row, int kbase, int half) {
    union { uint4 u[2]; v16h v; } cv;
    const uint4* p = (const uint4*)__builtin_assume_aligned(row + kbase + 8 * half, 16);
    cv.u[0] = p[0];
    cv.u[1] = p[2];  // +32 bytes (skip the other half's run)
    return cv.v;
}

// Layer-1 A fragments: f16 feature row (CX channels, stride CXP) followed by
// two relative-position channels at k==CX, CX+1; zero beyond.  Full 32-wide
// K blocks take the vector path (condition folds after unrolling).
template <int CX, int KT1>
__device__ __forceinline__ void load_features(v16h (&a1)[KT1], const _Float16* xrow,
                                              float rx, float ry, int half) {
#pragma unroll
    for (int kt = 0; kt < KT1; ++kt) {
        if (kt * 32 + 32 <= CX) {
            a1[kt] = load_frag16(xrow, kt * 32, half);
        } else {
#pragma unroll
            for (int e = 0; e < 16; ++e) {
                int k = kmap(e, half) + kt * 32;
                float v = 0.0f;
                if (k < CX)           v = (float)xrow[k];
                else if (k == CX)     v = rx;
                else if (k == CX + 1) v = ry;
                a1[kt][e] = (_Float16)v;
            }
        }
    }
}

// ---------------------------------------------------------------------------
// 0) Pre-pack kernels (run once per call; trivial cost, weights live in L2).
// ---------------------------------------------------------------------------
// f32 [K x N] row-major  ->  f16 [N x KP] (transposed, K zero-padded to KP).
__global__ __launch_bounds__(256) void pack_wT_kernel(const float* __restrict__ src,
                                                      _Float16* __restrict__ dst,
                                                      int K, int N, int KP) {
    int i = blockIdx.x * 256 + threadIdx.x;
    if (i >= N * KP) return;
    int n = i / KP, k = i - n * KP;
    dst[i] = (_Float16)((k < K) ? src[(size_t)k * N + n] : 0.0f);
}

// f32 [rows x C] -> f16 [rows x CP] zero-padded rows.
__global__ __launch_bounds__(256) void pack_rows_kernel(const float* __restrict__ src,
                                                        _Float16* __restrict__ dst,
                                                        int rows, int C, int CP) {
    int i = blockIdx.x * 256 + threadIdx.x;
    if (i >= rows * CP) return;
    int r = i / CP, c = i - r * CP;
    dst[i] = (_Float16)((c < C) ? src[(size_t)r * C + c] : 0.0f);
}

// ---------------------------------------------------------------------------
// 1) Farthest point sampling: one workgroup per cloud; pos + running min-d2
//    cached in LDS; block argmax with first-index tie-break (matches argmax).
// ---------------------------------------------------------------------------
__global__ __launch_bounds__(256) void fps_kernel(const float* __restrict__ pos,
                                                  int N, int M,
                                                  int* __restrict__ idx_out,
                                                  float* __restrict__ posq) {
    __shared__ float spx[4096];
    __shared__ float spy[4096];
    __shared__ float sd[4096];
    __shared__ float rv[256];
    __shared__ int   ri[256];

    const int b = blockIdx.x, t = threadIdx.x;
    const float* pb = pos + (size_t)b * N * 2;
    for (int j = t; j < N; j += 256) {
        spx[j] = pb[j * 2 + 0];
        spy[j] = pb[j * 2 + 1];
        sd[j]  = BIGF;
    }
    __syncthreads();

    int prev = 0;
    if (t == 0) {
        idx_out[(size_t)b * M] = 0;
        posq[(size_t)b * M * 2 + 0] = spx[0];
        posq[(size_t)b * M * 2 + 1] = spy[0];
    }
    for (int i = 1; i < M; ++i) {
        const float lx = spx[prev], ly = spy[prev];
        float bv = -1.0f;
        int   bi = N;
        for (int j = t; j < N; j += 256) {
            float dx = spx[j] - lx, dy = spy[j] - ly;
            float nd = fminf(sd[j], dx * dx + dy * dy);
            sd[j] = nd;
            if (nd > bv || (nd == bv && j < bi)) { bv = nd; bi = j; }
        }
        rv[t] = bv; ri[t] = bi;
        __syncthreads();
        for (int s = 128; s > 0; s >>= 1) {
            if (t < s) {
                float ov = rv[t + s]; int oi = ri[t + s];
                if (ov > rv[t] || (ov == rv[t] && oi < ri[t])) { rv[t] = ov; ri[t] = oi; }
            }
            __syncthreads();
        }
        prev = ri[0];
        if (t == 0) {
            idx_out[(size_t)b * M + i] = prev;
            posq[((size_t)b * M + i) * 2 + 0] = spx[prev];
            posq[((size_t)b * M + i) * 2 + 1] = spy[prev];
        }
        __syncthreads();
    }
}

// ---------------------------------------------------------------------------
// 2) Radius-limited 64-NN: one thread per query; 64-slot insertion buffer
//    striped across LDS columns (conflict-free).  key = (d2<=r2 ? d2 : BIG)
//    replicates top_k over where(d2<=r2, -d2, -BIG); validity -> bitmask.
// ---------------------------------------------------------------------------
__global__ __launch_bounds__(128) void nbr_kernel(const float* __restrict__ pos,
                                                  const float* __restrict__ posq,
                                                  int* __restrict__ nbr,
                                                  unsigned long long* __restrict__ vmask,
                                                  int N, int M, float r2) {
    __shared__ float sdist[64 * 128];
    __shared__ int   sidx[64 * 128];
    const int t = threadIdx.x;
    const int q = blockIdx.x * 128 + t;
    const int b = q / M;
    const float qx = posq[(size_t)q * 2 + 0];
    const float qy = posq[(size_t)q * 2 + 1];
    const float* pb = pos + (size_t)b * N * 2;

    int   cnt = 0, wslot = 0;
    float worst = -1.0f;
    for (int j = 0; j < N; ++j) {
        float dx = pb[j * 2 + 0] - qx, dy = pb[j * 2 + 1] - qy;
        float d2 = dx * dx + dy * dy;
        float key = (d2 <= r2) ? d2 : BIGF;
        if (cnt < 64) {
            sdist[cnt * 128 + t] = key; sidx[cnt * 128 + t] = j;
            if (key > worst) { worst = key; wslot = cnt; }
            ++cnt;
        } else if (key < worst) {
            sdist[wslot * 128 + t] = key; sidx[wslot * 128 + t] = j;
            worst = -1.0f;
            for (int k = 0; k < 64; ++k) {
                float v = sdist[k * 128 + t];
                if (v > worst) { worst = v; wslot = k; }
            }
        }
    }
    unsigned long long vm = 0ull;
    for (int k = 0; k < 64; ++k) {
        nbr[(size_t)q * 64 + k] = sidx[k * 128 + t];
        if (sdist[k * 128 + t] <= r2) vm |= (1ull << k);
    }
    vmask[q] = vm;
}

// ---------------------------------------------------------------------------
// 3) PointConv: one wave per query.  A = [64 nbrs x (CX+2)] gathered f16
//    features; layer1 WMMA -> ReLU -> LDS tile -> layer2 WMMA -> masked max.
//    Weights are pre-transposed f16 [NOUT x KPAD] -> B frags are 2x b128.
// ---------------------------------------------------------------------------
template <int CX, int CXP, int KP1, int CH, int COUT>
__global__ __launch_bounds__(256) void pointconv_kernel(
    const _Float16* __restrict__ x16, const float* __restrict__ pos,
    const float* __restrict__ posq, const int* __restrict__ nbr,
    const unsigned long long* __restrict__ vmask,
    const _Float16* __restrict__ waT, const float* __restrict__ ba,
    const _Float16* __restrict__ wbT, const float* __restrict__ bb,
    _Float16* __restrict__ out16, int N, int M) {
    constexpr int KT1 = KP1 / 32;
    constexpr int KT2 = CH / 32;
    __shared__ __align__(16) _Float16 Htile[8][16 * CH];

    const int w = threadIdx.x >> 5, l = threadIdx.x & 31;
    const int half = l >> 4, idx = l & 15;
    const int q = blockIdx.x * 8 + w;
    const int b = q / M;
    const float qx = posq[(size_t)q * 2 + 0];
    const float qy = posq[(size_t)q * 2 + 1];
    const int* nb = nbr + (size_t)q * 64;
    const unsigned long long vm = vmask[q];

    v16h a2[4][KT2];
#pragma unroll
    for (int mt = 0; mt < 4; ++mt) {
        __syncthreads();
        // Layer-1 A fragments for this row block (gathered from global f16).
        v16h a1[KT1];
        {
            const int m = mt * 16 + idx;
            const int j = nb[m];
            const _Float16* xrow = x16 + ((size_t)b * N + j) * CXP;
            const float rx = pos[((size_t)b * N + j) * 2 + 0] - qx;
            const float ry = pos[((size_t)b * N + j) * 2 + 1] - qy;
            load_features<CX, KT1>(a1, xrow, rx, ry, half);
        }
        for (int ntc = 0; ntc < CH / 16; ++ntc) {
            const int n = ntc * 16 + idx;
            v8f acc = {};
#pragma unroll
            for (int kt = 0; kt < KT1; ++kt)
                acc = wmma_f16(a1[kt], load_frag16(waT + (size_t)n * KP1, kt * 32, half), acc);
            const float bias = ba[n];
#pragma unroll
            for (int r = 0; r < 8; ++r) {
                float v = acc[r] + bias;
                v = v > 0.0f ? v : 0.0f;
                Htile[w][(r + 8 * half) * CH + n] = (_Float16)v;
            }
        }
        __syncthreads();
#pragma unroll
        for (int kt = 0; kt < KT2; ++kt)
            a2[mt][kt] = load_frag16(&Htile[w][idx * CH], kt * 32, half);
    }

    // Layer 2 + masked max over the 64 neighbor rows.
    for (int nto = 0; nto < COUT / 16; ++nto) {
        const int n = nto * 16 + idx;
        const float bias = bb[n];
        float mv = -BIGF;
#pragma unroll
        for (int mt = 0; mt < 4; ++mt) {
            v8f acc = {};
#pragma unroll
            for (int kt = 0; kt < KT2; ++kt)
                acc = wmma_f16(a2[mt][kt], load_frag16(wbT + (size_t)n * CH, kt * 32, half), acc);
#pragma unroll
            for (int r = 0; r < 8; ++r) {
                int m = mt * 16 + r + 8 * half;
                float v = acc[r] + bias;
                if ((vm >> m) & 1ull) mv = fmaxf(mv, v);
            }
        }
        mv = fmaxf(mv, __shfl_xor(mv, 16, 32));  // merge lane halves (same n)
        mv = fmaxf(mv, -60000.0f);               // keep representable in f16
        if (l < 16) out16[(size_t)q * COUT + n] = (_Float16)mv;
    }
}

// ---------------------------------------------------------------------------
// 4) Global SA: MLP([h2,p2]) per point + max over 512 points per cloud.
//    One block per cloud, one wave per 64-point group, block reduce in LDS.
// ---------------------------------------------------------------------------
__global__ __launch_bounds__(256) void sa3_kernel(const _Float16* __restrict__ h2,
                                                  const float* __restrict__ p2,
                                                  const _Float16* __restrict__ w3aT,
                                                  const float* __restrict__ b3a,
                                                  const _Float16* __restrict__ w3bT,
                                                  const float* __restrict__ b3b,
                                                  _Float16* __restrict__ g16) {
    constexpr int M = 512, KP1 = 160, KT1 = 5, CH = 128, KT2 = 4, COUT = 256;
    __shared__ __align__(16) _Float16 Htile[8][16 * CH];
    __shared__ float part[8][COUT];

    const int w = threadIdx.x >> 5, l = threadIdx.x & 31;
    const int half = l >> 4, idx = l & 15;
    const int b = blockIdx.x;
    const int base = w * 64;

    v16h a2[4][KT2];
#pragma unroll
    for (int mt = 0; mt < 4; ++mt) {
        __syncthreads();
        v16h a1[KT1];
        {
            const int m = base + mt * 16 + idx;
            const _Float16* hr = h2 + ((size_t)b * M + m) * 128;
            const float px = p2[((size_t)b * M + m) * 2 + 0];
            const float py = p2[((size_t)b * M + m) * 2 + 1];
            load_features<128, KT1>(a1, hr, px, py, half);
        }
        for (int ntc = 0; ntc < CH / 16; ++ntc) {
            const int n = ntc * 16 + idx;
            v8f acc = {};
#pragma unroll
            for (int kt = 0; kt < KT1; ++kt)
                acc = wmma_f16(a1[kt], load_frag16(w3aT + (size_t)n * KP1, kt * 32, half), acc);
            const float bias = b3a[n];
#pragma unroll
            for (int r = 0; r < 8; ++r) {
                float v = acc[r] + bias;
                v = v > 0.0f ? v : 0.0f;
                Htile[w][(r + 8 * half) * CH + n] = (_Float16)v;
            }
        }
        __syncthreads();
#pragma unroll
        for (int kt = 0; kt < KT2; ++kt)
            a2[mt][kt] = load_frag16(&Htile[w][idx * CH], kt * 32, half);
    }

    for (int nto = 0; nto < COUT / 16; ++nto) {
        const int n = nto * 16 + idx;
        const float bias = b3b[n];
        float mv = -3.0e38f;
#pragma unroll
        for (int mt = 0; mt < 4; ++mt) {
            v8f acc = {};
#pragma unroll
            for (int kt = 0; kt < KT2; ++kt)
                acc = wmma_f16(a2[mt][kt], load_frag16(w3bT + (size_t)n * CH, kt * 32, half), acc);
#pragma unroll
            for (int r = 0; r < 8; ++r) mv = fmaxf(mv, acc[r] + bias);
        }
        mv = fmaxf(mv, __shfl_xor(mv, 16, 32));
        if (l < 16) part[w][n] = mv;
    }
    __syncthreads();
    const int t = threadIdx.x;
    if (t < COUT) {
        float v = part[0][t];
#pragma unroll
        for (int ww = 1; ww < 8; ++ww) v = fmaxf(v, part[ww][t]);
        g16[(size_t)b * COUT + t] = (_Float16)v;
    }
}

// ---------------------------------------------------------------------------
// 5) Head MLP: [8,256] -> ReLU 256 -> 1024.  Single wave, rows padded to 16.
// ---------------------------------------------------------------------------
__global__ __launch_bounds__(32) void head_kernel(const _Float16* __restrict__ g16,
                                                  const _Float16* __restrict__ w4aT,
                                                  const float* __restrict__ b4a,
                                                  const _Float16* __restrict__ w4bT,
                                                  const float* __restrict__ b4b,
                                                  float* __restrict__ out) {
    __shared__ __align__(16) _Float16 Htile[16 * 256];
    const int l = threadIdx.x, half = l >> 4, idx = l & 15;

    v16h zf;
#pragma unroll
    for (int e = 0; e < 16; ++e) zf[e] = (_Float16)0.0f;

    v16h a1[8];
    {
        const _Float16* grow = g16 + (size_t)(idx & 7) * 256;  // always in-bounds
#pragma unroll
        for (int kt = 0; kt < 8; ++kt) {
            v16h t = load_frag16(grow, kt * 32, half);
            a1[kt] = (idx < 8) ? t : zf;  // pad rows 8..15 with zeros
        }
    }
    for (int nt = 0; nt < 16; ++nt) {
        const int n = nt * 16 + idx;
        v8f acc = {};
#pragma unroll
        for (int kt = 0; kt < 8; ++kt)
            acc = wmma_f16(a1[kt], load_frag16(w4aT + (size_t)n * 256, kt * 32, half), acc);
        const float bias = b4a[n];
#pragma unroll
        for (int r = 0; r < 8; ++r) {
            float v = acc[r] + bias;
            v = v > 0.0f ? v : 0.0f;
            Htile[(r + 8 * half) * 256 + n] = (_Float16)v;
        }
    }
    __syncthreads();
    v16h a2[8];
#pragma unroll
    for (int kt = 0; kt < 8; ++kt)
        a2[kt] = load_frag16(&Htile[idx * 256], kt * 32, half);

    for (int nt = 0; nt < 64; ++nt) {
        const int n = nt * 16 + idx;
        v8f acc = {};
#pragma unroll
        for (int kt = 0; kt < 8; ++kt)
            acc = wmma_f16(a2[kt], load_frag16(w4bT + (size_t)n * 256, kt * 32, half), acc);
        if (half == 0) {
            const float bias = b4b[n];
#pragma unroll
            for (int r = 0; r < 8; ++r)
                out[(size_t)r * 1024 + n] = acc[r] + bias;  // row r == batch r
        }
    }
}

// ---------------------------------------------------------------------------
// Launcher
// ---------------------------------------------------------------------------
extern "C" void kernel_launch(void* const* d_in, const int* in_sizes, int n_in,
                              void* d_out, int out_size, void* d_ws, size_t ws_size,
                              hipStream_t stream) {
    (void)in_sizes; (void)n_in; (void)out_size; (void)ws_size;
    const float* x   = (const float*)d_in[0];
    const float* pos = (const float*)d_in[1];
    const float* w1a = (const float*)d_in[2];
    const float* b1a = (const float*)d_in[3];
    const float* w1b = (const float*)d_in[4];
    const float* b1b = (const float*)d_in[5];
    const float* w2a = (const float*)d_in[6];
    const float* b2a = (const float*)d_in[7];
    const float* w2b = (const float*)d_in[8];
    const float* b2b = (const float*)d_in[9];
    const float* w3a = (const float*)d_in[10];
    const float* b3a = (const float*)d_in[11];
    const float* w3b = (const float*)d_in[12];
    const float* b3b = (const float*)d_in[13];
    const float* w4a = (const float*)d_in[14];
    const float* b4a = (const float*)d_in[15];
    const float* w4b = (const float*)d_in[16];
    const float* b4b = (const float*)d_in[17];
    float* out = (float*)d_out;

    const int B = 8, N1 = 4096, M1 = 2048, M2 = 512;

    char*  ws = (char*)d_ws;
    size_t o  = 0;
    auto alloc = [&](size_t bytes) -> void* {
        void* p = ws + o;
        o = (o + bytes + 255) & ~(size_t)255;
        return p;
    };
    int*   idx1 = (int*)alloc((size_t)B * M1 * 4);
    float* p1   = (float*)alloc((size_t)B * M1 * 2 * 4);
    int*   nbr1 = (int*)alloc((size_t)B * M1 * 64 * 4);
    unsigned long long* vm1 = (unsigned long long*)alloc((size_t)B * M1 * 8);
    int*   idx2 = (int*)alloc((size_t)B * M2 * 4);
    float* p2   = (float*)alloc((size_t)B * M2 * 2 * 4);
    int*   nbr2 = (int*)alloc((size_t)B * M2 * 64 * 4);
    unsigned long long* vm2 = (unsigned long long*)alloc((size_t)B * M2 * 8);

    _Float16* xf16 = (_Float16*)alloc((size_t)B * N1 * 8 * 2);
    _Float16* h1   = (_Float16*)alloc((size_t)B * M1 * 64 * 2);
    _Float16* h2   = (_Float16*)alloc((size_t)B * M2 * 128 * 2);
    _Float16* g16  = (_Float16*)alloc((size_t)B * 256 * 2);

    _Float16* w1aT = (_Float16*)alloc((size_t)32 * 32 * 2);
    _Float16* w1bT = (_Float16*)alloc((size_t)64 * 32 * 2);
    _Float16* w2aT = (_Float16*)alloc((size_t)64 * 96 * 2);
    _Float16* w2bT = (_Float16*)alloc((size_t)128 * 64 * 2);
    _Float16* w3aT = (_Float16*)alloc((size_t)128 * 160 * 2);
    _Float16* w3bT = (_Float16*)alloc((size_t)256 * 128 * 2);
    _Float16* w4aT = (_Float16*)alloc((size_t)256 * 256 * 2);
    _Float16* w4bT = (_Float16*)alloc((size_t)1024 * 256 * 2);

    auto gridFor = [](int elems) { return (elems + 255) / 256; };

    // --- Pre-pack operands to f16, K-contiguous ---
    pack_rows_kernel<<<gridFor(B * N1 * 8), 256, 0, stream>>>(x, xf16, B * N1, 5, 8);
    pack_wT_kernel<<<gridFor(32 * 32), 256, 0, stream>>>(w1a, w1aT, 7, 32, 32);
    pack_wT_kernel<<<gridFor(64 * 32), 256, 0, stream>>>(w1b, w1bT, 32, 64, 32);
    pack_wT_kernel<<<gridFor(64 * 96), 256, 0, stream>>>(w2a, w2aT, 66, 64, 96);
    pack_wT_kernel<<<gridFor(128 * 64), 256, 0, stream>>>(w2b, w2bT, 64, 128, 64);
    pack_wT_kernel<<<gridFor(128 * 160), 256, 0, stream>>>(w3a, w3aT, 130, 128, 160);
    pack_wT_kernel<<<gridFor(256 * 128), 256, 0, stream>>>(w3b, w3bT, 128, 256, 128);
    pack_wT_kernel<<<gridFor(256 * 256), 256, 0, stream>>>(w4a, w4aT, 256, 256, 256);
    pack_wT_kernel<<<gridFor(1024 * 256), 256, 0, stream>>>(w4b, w4bT, 256, 1024, 256);

    // --- SA1 ---
    fps_kernel<<<B, 256, 0, stream>>>(pos, N1, M1, idx1, p1);
    nbr_kernel<<<(B * M1) / 128, 128, 0, stream>>>(pos, p1, nbr1, vm1, N1, M1, 0.04f);
    pointconv_kernel<5, 8, 32, 32, 64><<<(B * M1) / 8, 256, 0, stream>>>(
        xf16, pos, p1, nbr1, vm1, w1aT, b1a, w1bT, b1b, h1, N1, M1);

    // --- SA2 ---
    fps_kernel<<<B, 256, 0, stream>>>(p1, M1, M2, idx2, p2);
    nbr_kernel<<<(B * M2) / 128, 128, 0, stream>>>(p1, p2, nbr2, vm2, M1, M2, 0.16f);
    pointconv_kernel<64, 64, 96, 64, 128><<<(B * M2) / 8, 256, 0, stream>>>(
        h1, p1, p2, nbr2, vm2, w2aT, b2a, w2bT, b2b, h2, M1, M2);

    // --- SA3 (global) + head ---
    sa3_kernel<<<B, 256, 0, stream>>>(h2, p2, w3aT, b3a, w3bT, b3b, g16);
    head_kernel<<<1, 32, 0, stream>>>(g16, w4aT, b4a, w4bT, b4b, out);
}
